// FeedForwardQuantum_65481071409701
// MI455X (gfx1250) — matrix operand order repaired and checked by reading, not compiled, not run on hardware
//
#include <hip/hip_runtime.h>

#define MTOT  32768   // B*S
#define EMB   1024
#define FFN_  4096
#define NQ_   8
#define ROWH  40      // LDS tile row length in halves (32 data + 8 pad) = 80B, 16B-aligned

typedef __attribute__((ext_vector_type(16))) __bf16       v16bf;
typedef __attribute__((ext_vector_type(8)))  float        v8f;
typedef __attribute__((ext_vector_type(4)))  unsigned int u32x4;
typedef unsigned long long ull;

union AFrag { v16bf bf; u32x4 q[2]; };

__device__ __forceinline__ unsigned short f2bf(float f) {
  unsigned u = __builtin_bit_cast(unsigned, f);
  u += 0x7FFFu + ((u >> 16) & 1u);          // round-to-nearest-even
  return (unsigned short)(u >> 16);
}
__device__ __forceinline__ unsigned pack2bf(float a, float b) {
  return (unsigned)f2bf(a) | ((unsigned)f2bf(b) << 16);
}

// async copy 16B global -> LDS (ASYNCcnt-tracked, bypasses VGPRs)
__device__ __forceinline__ void async_g2l_b128(unsigned ldsOff, const void* gaddr) {
  asm volatile("global_load_async_to_lds_b128 %0, %1, off"
               :: "v"(ldsOff), "v"((ull)gaddr) : "memory");
}
__device__ __forceinline__ void wait_async6() {
#if __has_builtin(__builtin_amdgcn_s_wait_asynccnt)
  __builtin_amdgcn_s_wait_asynccnt(6);
#else
  asm volatile("s_wait_asynccnt 6" ::: "memory");
#endif
}
__device__ __forceinline__ void wait_async0() {
#if __has_builtin(__builtin_amdgcn_s_wait_asynccnt)
  __builtin_amdgcn_s_wait_asynccnt(0);
#else
  asm volatile("s_wait_asynccnt 0" ::: "memory");
#endif
}

// ---- W2 (f32, [EMB, FFN]) -> bf16 ----
__global__ void __launch_bounds__(256)
conv_w2_kernel(const float* __restrict__ W2, unsigned short* __restrict__ w2bf) {
  int i = (blockIdx.x * 256 + threadIdx.x) * 4;
  const float4 v = *(const float4*)(W2 + i);
  uint2 p; p.x = pack2bf(v.x, v.y); p.y = pack2bf(v.z, v.w);
  *(uint2*)(w2bf + i) = p;
}

// ---- W1' = W1 * cos(theta) -> bf16, [FFN, 8] ----
__global__ void __launch_bounds__(256)
conv_w1_kernel(const float* __restrict__ W1, const float* __restrict__ theta,
               unsigned short* __restrict__ w1bf) {
  int idx = blockIdx.x * 256 + threadIdx.x;           // 32768 elements
  w1bf[idx] = f2bf(W1[idx] * __builtin_cosf(theta[idx & 7]));
}

// ---- h = relu(cos(x[:, :8]) @ W1'^T + b1) via WMMA with K padded 8->32 ----
__global__ void __launch_bounds__(256)
h_wmma_kernel(const float* __restrict__ x, const unsigned short* __restrict__ w1bf,
              const float* __restrict__ b1, unsigned short* __restrict__ h) {
  const int tid  = threadIdx.x;
  const int lane = tid & 31;
  const int wave = tid >> 5;
  const int hg   = lane >> 4;
  const int l15  = lane & 15;
  const int m0   = blockIdx.y * 128 + wave * 16;      // wave owns a 16-row m-tile
  const int f0   = blockIdx.x * 512;                  // 32 f-tiles of 16

  const u32x4 z = {0u, 0u, 0u, 0u};
  AFrag a; a.q[1] = z;
  if (lane < 16) {
    const float* xr = x + (size_t)(m0 + l15) * EMB;
    float4 lo = *(const float4*)xr;
    float4 hi = *(const float4*)(xr + 4);
    a.q[0] = (u32x4){pack2bf(__builtin_cosf(lo.x), __builtin_cosf(lo.y)),
                     pack2bf(__builtin_cosf(lo.z), __builtin_cosf(lo.w)),
                     pack2bf(__builtin_cosf(hi.x), __builtin_cosf(hi.y)),
                     pack2bf(__builtin_cosf(hi.z), __builtin_cosf(hi.w))};
  } else {
    a.q[0] = z;
  }

  #pragma unroll 4
  for (int t = 0; t < 32; t++) {
    const int fc = f0 + t * 16;
    AFrag b; b.q[1] = z;
    if (lane < 16) b.q[0] = *(const u32x4*)(w1bf + (size_t)(fc + l15) * NQ_);
    else           b.q[0] = z;

    v8f acc = (v8f){0.f, 0.f, 0.f, 0.f, 0.f, 0.f, 0.f, 0.f};
    acc = __builtin_amdgcn_wmma_f32_16x16x32_bf16(
        false, a.bf, false, b.bf, (short)0, acc, false, false);

    const int col = fc + l15;
    const float bb = b1[col];
    #pragma unroll
    for (int j = 0; j < 8; j++) {
      float s = fmaxf(acc[j] + bb, 0.0f);
      h[(size_t)(m0 + j + 8 * hg) * FFN_ + col] = f2bf(s);
    }
  }
}

// ---- out = h @ W2^T + b2 : bf16 WMMA GEMM ----
// WG = 128 rows x 256 cols, 8 waves each 64x64 (4x4 WMMA tiles).
// Double-buffered async global->LDS staging (ASYNCcnt) overlapped with WMMA.
__global__ void __launch_bounds__(256)
gemm2_kernel(const unsigned short* __restrict__ h, const unsigned short* __restrict__ w2bf,
             const float* __restrict__ b2, float* __restrict__ out) {
  __shared__ unsigned short aT[2][128 * ROWH];
  __shared__ unsigned short bT[2][256 * ROWH];

  const int tid  = threadIdx.x;
  const int lane = tid & 31;
  const int wave = tid >> 5;
  const int wm   = wave >> 2;          // 0..1 : 64-row block
  const int wn   = wave & 3;           // 0..3 : 64-col block
  const int hg   = lane >> 4;
  const int l15  = lane & 15;
  const int m0   = blockIdx.y * 128;
  const int e0   = blockIdx.x * 256;

  v8f acc[4][4];
  #pragma unroll
  for (int tm = 0; tm < 4; tm++)
    #pragma unroll
    for (int tn = 0; tn < 4; tn++)
      acc[tm][tn] = (v8f){0.f, 0.f, 0.f, 0.f, 0.f, 0.f, 0.f, 0.f};

  // stage one K-slab (A: 128x32, B: 256x32 halves) into buffer `buf`
  auto stage = [&](int buf, int k0) {
    #pragma unroll
    for (int c = tid; c < 1536; c += 256) {            // 6 chunks / thread
      if (c < 512) {
        const int n = c >> 2, ch = c & 3;
        const unsigned short* src = h + (size_t)(m0 + n) * FFN_ + k0 + ch * 8;
        async_g2l_b128((unsigned)(uintptr_t)&aT[buf][n * ROWH + ch * 8], src);
      } else {
        const int cb = c - 512;
        const int n = cb >> 2, ch = cb & 3;
        const unsigned short* src = w2bf + (size_t)(e0 + n) * FFN_ + k0 + ch * 8;
        async_g2l_b128((unsigned)(uintptr_t)&bT[buf][n * ROWH + ch * 8], src);
      }
    }
  };

  stage(0, 0);
  int cur = 0;
  for (int k0 = 0; k0 < FFN_; k0 += 32, cur ^= 1) {
    if (k0 + 32 < FFN_) {
      stage(cur ^ 1, k0 + 32);   // prefetch next slab into other buffer
      wait_async6();             // in-order: current slab (oldest 6) complete
    } else {
      wait_async0();
    }
    __syncthreads();

    AFrag a[4], b[4];
    #pragma unroll
    for (int tm = 0; tm < 4; tm++) {
      const int off = (wm * 64 + tm * 16 + l15) * ROWH + hg * 8;
      a[tm].q[0] = *(const u32x4*)(&aT[cur][off]);
      a[tm].q[1] = *(const u32x4*)(&aT[cur][off + 16]);
    }
    #pragma unroll
    for (int tn = 0; tn < 4; tn++) {
      const int off = (wn * 64 + tn * 16 + l15) * ROWH + hg * 8;
      b[tn].q[0] = *(const u32x4*)(&bT[cur][off]);
      b[tn].q[1] = *(const u32x4*)(&bT[cur][off + 16]);
    }
    #pragma unroll
    for (int tm = 0; tm < 4; tm++)
      #pragma unroll
      for (int tn = 0; tn < 4; tn++)
        acc[tm][tn] = __builtin_amdgcn_wmma_f32_16x16x32_bf16(
            false, a[tm].bf, false, b[tn].bf, (short)0, acc[tm][tn], false, false);
    __syncthreads();   // all frag reads of `cur` done before it is re-targeted
  }

  #pragma unroll
  for (int tn = 0; tn < 4; tn++) {
    const int col = e0 + wn * 64 + tn * 16 + l15;
    const float bb = b2[col];
    #pragma unroll
    for (int tm = 0; tm < 4; tm++) {
      const int rbase = m0 + wm * 64 + tm * 16 + hg * 8;
      #pragma unroll
      for (int j = 0; j < 8; j++)
        out[(size_t)(rbase + j) * EMB + col] = acc[tm][tn][j] + bb;
    }
  }
}

// ---- fallback (small ws): fused kernel, VALU-recomputes h tiles ----
__global__ void __launch_bounds__(256)
ffq_fused_kernel(const float* __restrict__ x, const float* __restrict__ theta,
                 const float* __restrict__ W1, const float* __restrict__ b1,
                 const unsigned short* __restrict__ w2bf, const float* __restrict__ b2,
                 float* __restrict__ out) {
  __shared__ unsigned short aTile[128 * ROWH];
  __shared__ unsigned short bTile[128 * ROWH];
  __shared__ float qs[128 * NQ_];
  __shared__ float cth[NQ_];

  const int tid  = threadIdx.x;
  const int lane = tid & 31;
  const int wave = tid >> 5;
  const int wm   = wave >> 2;
  const int wn   = wave & 3;
  const int hg   = lane >> 4;
  const int l15  = lane & 15;
  const int m0   = blockIdx.y * 128;
  const int e0   = blockIdx.x * 128;

  if (tid < NQ_) cth[tid] = __builtin_cosf(theta[tid]);
  __syncthreads();
  for (int idx = tid; idx < 128 * NQ_; idx += 256) {
    const int m = idx >> 3, i = idx & 7;
    qs[idx] = __builtin_cosf(x[(size_t)(m0 + m) * EMB + i]) * cth[i];
  }
  __syncthreads();

  const int am = tid >> 1;
  const int fb = (tid & 1) * 16;
  float q[NQ_];
  #pragma unroll
  for (int i = 0; i < NQ_; i++) q[i] = qs[am * NQ_ + i];

  v8f acc[4][2];
  #pragma unroll
  for (int tm = 0; tm < 4; tm++)
    #pragma unroll
    for (int tn = 0; tn < 2; tn++)
      acc[tm][tn] = (v8f){0.f, 0.f, 0.f, 0.f, 0.f, 0.f, 0.f, 0.f};

  for (int k0 = 0; k0 < FFN_; k0 += 32) {
    unsigned pk[8];
    #pragma unroll
    for (int j = 0; j < 16; j += 2) {
      float sv[2];
      #pragma unroll
      for (int jj = 0; jj < 2; jj++) {
        const int f = k0 + fb + j + jj;
        const float* w1r = W1 + (size_t)f * NQ_;
        float s = b1[f];
        #pragma unroll
        for (int i = 0; i < NQ_; i++) s += q[i] * w1r[i];
        sv[jj] = fmaxf(s, 0.0f);
      }
      pk[j >> 1] = pack2bf(sv[0], sv[1]);
    }
    *(u32x4*)(&aTile[am * ROWH + fb])     = (u32x4){pk[0], pk[1], pk[2], pk[3]};
    *(u32x4*)(&aTile[am * ROWH + fb + 8]) = (u32x4){pk[4], pk[5], pk[6], pk[7]};

    #pragma unroll
    for (int c = tid; c < 512; c += 256) {
      const int n = c >> 2, ch = c & 3;
      const unsigned short* src = w2bf + (size_t)(e0 + n) * FFN_ + k0 + ch * 8;
      *(u32x4*)(&bTile[n * ROWH + ch * 8]) = *(const u32x4*)src;
    }
    __syncthreads();

    AFrag a[4], b[2];
    #pragma unroll
    for (int tm = 0; tm < 4; tm++) {
      const int off = (wm * 64 + tm * 16 + l15) * ROWH + hg * 8;
      a[tm].q[0] = *(const u32x4*)(&aTile[off]);
      a[tm].q[1] = *(const u32x4*)(&aTile[off + 16]);
    }
    #pragma unroll
    for (int tn = 0; tn < 2; tn++) {
      const int off = (wn * 32 + tn * 16 + l15) * ROWH + hg * 8;
      b[tn].q[0] = *(const u32x4*)(&bTile[off]);
      b[tn].q[1] = *(const u32x4*)(&bTile[off + 16]);
    }
    #pragma unroll
    for (int tm = 0; tm < 4; tm++)
      #pragma unroll
      for (int tn = 0; tn < 2; tn++)
        acc[tm][tn] = __builtin_amdgcn_wmma_f32_16x16x32_bf16(
            false, a[tm].bf, false, b[tn].bf, (short)0, acc[tm][tn], false, false);
    __syncthreads();
  }

  #pragma unroll
  for (int tn = 0; tn < 2; tn++) {
    const int col = e0 + wn * 32 + tn * 16 + l15;
    const float bb = b2[col];
    #pragma unroll
    for (int tm = 0; tm < 4; tm++) {
      const int rbase = m0 + wm * 64 + tm * 16 + hg * 8;
      #pragma unroll
      for (int j = 0; j < 8; j++)
        out[(size_t)(rbase + j) * EMB + col] = acc[tm][tn][j] + bb;
    }
  }
}

extern "C" void kernel_launch(void* const* d_in, const int* in_sizes, int n_in,
                              void* d_out, int out_size, void* d_ws, size_t ws_size,
                              hipStream_t stream) {
  const float* x     = (const float*)d_in[0];
  const float* theta = (const float*)d_in[1];
  const float* W1    = (const float*)d_in[2];
  const float* b1    = (const float*)d_in[3];
  const float* W2    = (const float*)d_in[4];
  const float* b2    = (const float*)d_in[5];
  float* out = (float*)d_out;

  const size_t w2Bytes = (size_t)EMB * FFN_ * 2;         // 8 MB
  const size_t w1Off   = w2Bytes;
  const size_t hOff    = w2Bytes + 65536;                // w1bf is exactly 64 KB
  const size_t need    = hOff + (size_t)MTOT * FFN_ * 2; // + 256 MB

  unsigned short* w2bf = (unsigned short*)d_ws;
  conv_w2_kernel<<<(EMB * FFN_) / (256 * 4), 256, 0, stream>>>(W2, w2bf);

  if (ws_size >= need) {
    unsigned short* w1bf = (unsigned short*)((char*)d_ws + w1Off);
    unsigned short* hbuf = (unsigned short*)((char*)d_ws + hOff);
    conv_w1_kernel<<<(FFN_ * NQ_) / 256, 256, 0, stream>>>(W1, theta, w1bf);
    h_wmma_kernel<<<dim3(FFN_ / 512, MTOT / 128), 256, 0, stream>>>(x, w1bf, b1, hbuf);
    gemm2_kernel<<<dim3(EMB / 256, MTOT / 128), 256, 0, stream>>>(hbuf, w2bf, b2, out);
  } else {
    ffq_fused_kernel<<<dim3(EMB / 128, MTOT / 128), 256, 0, stream>>>(x, theta, W1, b1, w2bf, b2, out);
  }
}
